// WindowAttention_68247030334216
// MI455X (gfx1250) — matrix operand order repaired and checked by reading, not compiled, not run on hardware
//
#include <hip/hip_runtime.h>

typedef __attribute__((ext_vector_type(16))) __bf16 v16bf;
typedef __attribute__((ext_vector_type(8)))  __bf16 v8bf;
typedef __attribute__((ext_vector_type(8)))  float  v8f;

#define NT     49      // tokens per window
#define C_DIM  384
#define H_DIM  12
#define HD     32
#define NW_    64
#define BW_    4096
#define M_TOT  (BW_ * NT)     // 200704
#define QKV_N  (3 * C_DIM)    // 1152

static __device__ __forceinline__ __bf16 f2bf(float f) { return (__bf16)f; }

// A-fragment (16x32 bf16, MxK): lane%16 = row M; lane/16 selects K-chunks:
// half=0 -> K in [0..7] and [16..23]; half=1 -> K in [8..15] and [24..31]
static __device__ __forceinline__ v16bf frag_a(const __bf16* row, int half) {
    v8bf lo = *(const v8bf*)(row + half * 8);
    v8bf hi = *(const v8bf*)(row + 16 + half * 8);
    return __builtin_shufflevector(lo, hi, 0,1,2,3,4,5,6,7,8,9,10,11,12,13,14,15);
}
// B-fragment (32x16 bf16, KxN) from K-contiguous storage per column:
// lane%16 = col N; lane/16 selects K-chunks [half*8..half*8+7] and [16+half*8..16+half*8+7]
static __device__ __forceinline__ v16bf frag_b(const __bf16* col, int half) {
    v8bf lo = *(const v8bf*)(col + half * 8);
    v8bf hi = *(const v8bf*)(col + 16 + half * 8);
    return __builtin_shufflevector(lo, hi, 0,1,2,3,4,5,6,7,8,9,10,11,12,13,14,15);
}

// ---------------------------------------------------------------------------
// Kernel 0: weight conversion + relative-position-bias gather
// ---------------------------------------------------------------------------
__global__ void prep_kernel(const float* __restrict__ qkv_w, const float* __restrict__ proj_w,
                            const float* __restrict__ bias_table, const int* __restrict__ rel_index,
                            __bf16* __restrict__ qkv_wb, __bf16* __restrict__ proj_wb,
                            float* __restrict__ biasHNN) {
    int id = blockIdx.x * 256 + threadIdx.x;
    if (id < QKV_N * C_DIM) qkv_wb[id] = f2bf(qkv_w[id]);
    if (id < C_DIM * C_DIM) proj_wb[id] = f2bf(proj_w[id]);
    if (id < H_DIM * NT * NT) {
        int h = id / (NT * NT);
        int r = id % (NT * NT);
        biasHNN[id] = bias_table[rel_index[r] * H_DIM + h];
    }
}

// ---------------------------------------------------------------------------
// Kernel 1: QKV projection GEMM (M=200704, N=1152, K=384), bf16 WMMA.
// Block tile 128x128, 8 waves of 32x64 (2x4 WMMA accumulators),
// double-buffered LDS, pipelined global loads, fragments hoisted so the
// scheduler can use partial DScnt waits.
// ---------------------------------------------------------------------------
__global__ __launch_bounds__(256) void qkv_gemm_kernel(
        const float* __restrict__ x, const __bf16* __restrict__ wb,
        const float* __restrict__ qkv_b, __bf16* __restrict__ qkv_out) {
    __shared__ __bf16 Alds[2][128][48];
    __shared__ __bf16 Blds[2][128][48];
    const int t = threadIdx.x;
    const int lane = t & 31, wave = t >> 5;
    const int wm = wave & 3, wn = wave >> 2;
    const int m0 = blockIdx.x * 128;
    const int n0 = blockIdx.y * 128;
    const int half = lane >> 4, jj = lane & 15;
    const int srow = t >> 1;
    const int scol = (t & 1) * 16;

    const float*  aptr = x  + (size_t)(m0 + srow) * C_DIM + scol;
    const __bf16* bptr = wb + (size_t)(n0 + srow) * C_DIM + scol;

    v8f zero = 0.0f;
    v8f acc[2][4];
    #pragma unroll
    for (int it = 0; it < 2; ++it)
      #pragma unroll
      for (int jt = 0; jt < 4; ++jt) acc[it][jt] = zero;

    float4 ar[4];
    uint4  br[2];
    {   // prefetch K-step 0
        const float4* ap4 = (const float4*)aptr;
        ar[0] = ap4[0]; ar[1] = ap4[1]; ar[2] = ap4[2]; ar[3] = ap4[3];
        const uint4* bp4 = (const uint4*)bptr;
        br[0] = bp4[0]; br[1] = bp4[1];
        __bf16* ad = &Alds[0][srow][scol];
        ad[0]  = f2bf(ar[0].x); ad[1]  = f2bf(ar[0].y); ad[2]  = f2bf(ar[0].z); ad[3]  = f2bf(ar[0].w);
        ad[4]  = f2bf(ar[1].x); ad[5]  = f2bf(ar[1].y); ad[6]  = f2bf(ar[1].z); ad[7]  = f2bf(ar[1].w);
        ad[8]  = f2bf(ar[2].x); ad[9]  = f2bf(ar[2].y); ad[10] = f2bf(ar[2].z); ad[11] = f2bf(ar[2].w);
        ad[12] = f2bf(ar[3].x); ad[13] = f2bf(ar[3].y); ad[14] = f2bf(ar[3].z); ad[15] = f2bf(ar[3].w);
        uint4* bd = (uint4*)&Blds[0][srow][scol];
        bd[0] = br[0]; bd[1] = br[1];
    }

    for (int ks = 0; ks < 12; ++ks) {
        const int cur = ks & 1, nxt = cur ^ 1;
        __syncthreads();
        if (ks < 11) {   // issue next-step global loads (no wait yet)
            const float4* ap4 = (const float4*)(aptr + (ks + 1) * 32);
            ar[0] = ap4[0]; ar[1] = ap4[1]; ar[2] = ap4[2]; ar[3] = ap4[3];
            const uint4* bp4 = (const uint4*)(bptr + (ks + 1) * 32);
            br[0] = bp4[0]; br[1] = bp4[1];
        }
        // load ALL fragments first so ds_loads can issue as one clause
        v16bf af0 = frag_a(&Alds[cur][wm * 32 + jj][0], half);
        v16bf af1 = frag_a(&Alds[cur][wm * 32 + 16 + jj][0], half);
        v16bf bfr[4];
        #pragma unroll
        for (int jt = 0; jt < 4; ++jt)
            bfr[jt] = frag_b(&Blds[cur][wn * 64 + jt * 16 + jj][0], half);
        #pragma unroll
        for (int jt = 0; jt < 4; ++jt) {
            acc[0][jt] = __builtin_amdgcn_wmma_f32_16x16x32_bf16(false, af0, false, bfr[jt], (short)0, acc[0][jt], false, false);
            acc[1][jt] = __builtin_amdgcn_wmma_f32_16x16x32_bf16(false, af1, false, bfr[jt], (short)0, acc[1][jt], false, false);
        }
        if (ks < 11) {   // convert + stage into the other buffer
            __bf16* ad = &Alds[nxt][srow][scol];
            ad[0]  = f2bf(ar[0].x); ad[1]  = f2bf(ar[0].y); ad[2]  = f2bf(ar[0].z); ad[3]  = f2bf(ar[0].w);
            ad[4]  = f2bf(ar[1].x); ad[5]  = f2bf(ar[1].y); ad[6]  = f2bf(ar[1].z); ad[7]  = f2bf(ar[1].w);
            ad[8]  = f2bf(ar[2].x); ad[9]  = f2bf(ar[2].y); ad[10] = f2bf(ar[2].z); ad[11] = f2bf(ar[2].w);
            ad[12] = f2bf(ar[3].x); ad[13] = f2bf(ar[3].y); ad[14] = f2bf(ar[3].z); ad[15] = f2bf(ar[3].w);
            uint4* bd = (uint4*)&Blds[nxt][srow][scol];
            bd[0] = br[0]; bd[1] = br[1];
        }
    }

    // epilogue: add bias, scatter bf16 into [s][b][h][n][d]
    #pragma unroll
    for (int it = 0; it < 2; ++it)
      #pragma unroll
      for (int jt = 0; jt < 4; ++jt)
        #pragma unroll
        for (int r = 0; r < 8; ++r) {
            int m = m0 + wm * 32 + it * 16 + half * 8 + r;
            int n = n0 + wn * 64 + jt * 16 + jj;
            float v = acc[it][jt][r] + qkv_b[n];
            int s  = n / C_DIM, rem = n % C_DIM;
            int h  = rem >> 5,  d   = rem & 31;
            int b  = m / NT,    nn  = m % NT;
            size_t off = (((size_t)s * BW_ + b) * H_DIM + h) * (NT * HD) + (size_t)nn * HD + d;
            qkv_out[off] = f2bf(v);
        }
}

// ---------------------------------------------------------------------------
// Kernel 2: fused attention per (window b, head h): S=qK^T, +bias+mask,
// softmax, O=PV. 128 threads = 4 waves; wave w owns S/O rows [16w,16w+16).
// q/k staged with GLOBAL_LOAD_ASYNC_TO_LDS_B128 (ASYNCcnt path).
// ---------------------------------------------------------------------------
__global__ __launch_bounds__(128) void attn_kernel(
        const __bf16* __restrict__ qkv, const float* __restrict__ mask,
        const float* __restrict__ biasHNN, __bf16* __restrict__ attn_out) {
    __shared__ __bf16 Qs[64][48];
    __shared__ __bf16 Ks[64][48];
    __shared__ __bf16 VT[32][80];   // v transposed: VT[d][token]
    __shared__ __bf16 Ps[64][72];

    const int b = blockIdx.x, h = blockIdx.y;
    const int t = threadIdx.x, lane = t & 31, wave = t >> 5;
    const int half = lane >> 4, jj = lane & 15;

    const size_t stride_s = (size_t)BW_ * H_DIM * NT * HD;
    const __bf16* qg = qkv + ((size_t)b * H_DIM + h) * (NT * HD);
    const __bf16* kg = qg + stride_s;
    const __bf16* vg = qg + 2 * stride_s;

    {   // stage q,k (async copies; zero-pad rows 49..63) and v transposed
        int row = t >> 1, col = (t & 1) * 16;
        if (row < NT) {
            unsigned q0 = (unsigned)(uintptr_t)&Qs[row][col];
            unsigned q1 = (unsigned)(uintptr_t)&Qs[row][col + 8];
            unsigned k0 = (unsigned)(uintptr_t)&Ks[row][col];
            unsigned k1 = (unsigned)(uintptr_t)&Ks[row][col + 8];
            const __bf16* qsrc = qg + row * HD + col;
            const __bf16* ksrc = kg + row * HD + col;
            asm volatile("global_load_async_to_lds_b128 %0, %1, off" :: "v"(q0), "v"(qsrc) : "memory");
            asm volatile("global_load_async_to_lds_b128 %0, %1, off" :: "v"(q1), "v"(qsrc + 8) : "memory");
            asm volatile("global_load_async_to_lds_b128 %0, %1, off" :: "v"(k0), "v"(ksrc) : "memory");
            asm volatile("global_load_async_to_lds_b128 %0, %1, off" :: "v"(k1), "v"(ksrc + 8) : "memory");
            #pragma unroll
            for (int j = 0; j < 16; ++j) VT[col + j][row] = vg[row * HD + col + j];
        } else {
            uint4 z = make_uint4(0u, 0u, 0u, 0u);
            *(uint4*)&Qs[row][col] = z; *(uint4*)(&Qs[row][col] + 8) = z;
            *(uint4*)&Ks[row][col] = z; *(uint4*)(&Ks[row][col] + 8) = z;
            #pragma unroll
            for (int j = 0; j < 16; ++j) VT[col + j][row] = (__bf16)0.0f;
        }
        asm volatile("s_wait_asynccnt 0x0" ::: "memory");
    }
    __syncthreads();

    // S = q k^T  (K = 32 = one WMMA step); hoist all fragments first
    v8f zero = 0.0f;
    v8f sacc[4]; sacc[0] = zero; sacc[1] = zero; sacc[2] = zero; sacc[3] = zero;
    v16bf aq = frag_a(&Qs[wave * 16 + jj][0], half);
    v16bf bk[4];
    #pragma unroll
    for (int jt = 0; jt < 4; ++jt) bk[jt] = frag_b(&Ks[jt * 16 + jj][0], half);
    #pragma unroll
    for (int jt = 0; jt < 4; ++jt)
        sacc[jt] = __builtin_amdgcn_wmma_f32_16x16x32_bf16(false, aq, false, bk[jt], (short)0, sacc[jt], false, false);

    // softmax rows, fully in registers (C-layout: row = half*8+r, col = jj)
    const float scale = 0.1767766952966369f;   // 1/sqrt(32)
    const float* mrow = mask + (size_t)(b & (NW_ - 1)) * NT * NT;
    const float* brow = biasHNN + (size_t)h * NT * NT;
    #pragma unroll
    for (int r = 0; r < 8; ++r) {
        int i = wave * 16 + half * 8 + r;
        float xv[4];
        #pragma unroll
        for (int jt = 0; jt < 4; ++jt) {
            int j = jt * 16 + jj;
            float add = (i < NT && j < NT) ? (mrow[i * NT + j] + brow[i * NT + j]) : -1e30f;
            xv[jt] = sacc[jt][r] * scale + add;
        }
        float mx = fmaxf(fmaxf(xv[0], xv[1]), fmaxf(xv[2], xv[3]));
        #pragma unroll
        for (int sh = 1; sh < 16; sh <<= 1) mx = fmaxf(mx, __shfl_xor(mx, sh, 32));
        float sum = 0.0f;
        #pragma unroll
        for (int jt = 0; jt < 4; ++jt) { xv[jt] = __expf(xv[jt] - mx); sum += xv[jt]; }
        #pragma unroll
        for (int sh = 1; sh < 16; sh <<= 1) sum += __shfl_xor(sum, sh, 32);
        float inv = 1.0f / (sum + 1e-20f);
        #pragma unroll
        for (int jt = 0; jt < 4; ++jt) Ps[i][jt * 16 + jj] = f2bf(xv[jt] * inv);
    }
    __syncthreads();

    // O = P (64x64) * V (64x32): 2 K-steps, 2 d-tiles
    v8f oacc[2]; oacc[0] = zero; oacc[1] = zero;
    #pragma unroll
    for (int ksp = 0; ksp < 2; ++ksp) {
        v16bf ap = frag_a(&Ps[wave * 16 + jj][ksp * 32], half);
        v16bf bv0 = frag_b(&VT[jj][ksp * 32], half);
        v16bf bv1 = frag_b(&VT[16 + jj][ksp * 32], half);
        oacc[0] = __builtin_amdgcn_wmma_f32_16x16x32_bf16(false, ap, false, bv0, (short)0, oacc[0], false, false);
        oacc[1] = __builtin_amdgcn_wmma_f32_16x16x32_bf16(false, ap, false, bv1, (short)0, oacc[1], false, false);
    }
    #pragma unroll
    for (int jt = 0; jt < 2; ++jt)
      #pragma unroll
      for (int r = 0; r < 8; ++r) {
          int i = wave * 16 + half * 8 + r;
          if (i < NT) {
              int d = jt * 16 + jj;
              size_t off = ((size_t)b * NT + i) * C_DIM + h * HD + d;
              attn_out[off] = f2bf(oacc[jt][r]);
          }
      }
}

// ---------------------------------------------------------------------------
// Kernel 3: output projection GEMM (M=200704, N=384, K=384), fp32 out.
// Block tile 128x128, double-buffered + pipelined like kernel 1.
// ---------------------------------------------------------------------------
__global__ __launch_bounds__(256) void proj_gemm_kernel(
        const __bf16* __restrict__ a, const __bf16* __restrict__ wb,
        const float* __restrict__ bias, float* __restrict__ out) {
    __shared__ __bf16 Alds[2][128][48];
    __shared__ __bf16 Blds[2][128][48];
    const int t = threadIdx.x;
    const int lane = t & 31, wave = t >> 5;
    const int wm = wave & 3, wn = wave >> 2;
    const int m0 = blockIdx.x * 128;
    const int n0 = blockIdx.y * 128;
    const int half = lane >> 4, jj = lane & 15;
    const int srow = t >> 1;
    const int scol = (t & 1) * 16;

    const __bf16* aptr = a  + (size_t)(m0 + srow) * C_DIM + scol;
    const __bf16* bptr = wb + (size_t)(n0 + srow) * C_DIM + scol;

    v8f zero = 0.0f;
    v8f acc[2][4];
    #pragma unroll
    for (int it = 0; it < 2; ++it)
      #pragma unroll
      for (int jt = 0; jt < 4; ++jt) acc[it][jt] = zero;

    uint4 ar[2], br[2];
    {   // prefetch K-step 0
        const uint4* ap4 = (const uint4*)aptr;
        ar[0] = ap4[0]; ar[1] = ap4[1];
        const uint4* bp4 = (const uint4*)bptr;
        br[0] = bp4[0]; br[1] = bp4[1];
        uint4* ad = (uint4*)&Alds[0][srow][scol];
        ad[0] = ar[0]; ad[1] = ar[1];
        uint4* bd = (uint4*)&Blds[0][srow][scol];
        bd[0] = br[0]; bd[1] = br[1];
    }

    for (int ks = 0; ks < 12; ++ks) {
        const int cur = ks & 1, nxt = cur ^ 1;
        __syncthreads();
        if (ks < 11) {
            const uint4* ap4 = (const uint4*)(aptr + (ks + 1) * 32);
            ar[0] = ap4[0]; ar[1] = ap4[1];
            const uint4* bp4 = (const uint4*)(bptr + (ks + 1) * 32);
            br[0] = bp4[0]; br[1] = bp4[1];
        }
        v16bf af0 = frag_a(&Alds[cur][wm * 32 + jj][0], half);
        v16bf af1 = frag_a(&Alds[cur][wm * 32 + 16 + jj][0], half);
        v16bf bfr[4];
        #pragma unroll
        for (int jt = 0; jt < 4; ++jt)
            bfr[jt] = frag_b(&Blds[cur][wn * 64 + jt * 16 + jj][0], half);
        #pragma unroll
        for (int jt = 0; jt < 4; ++jt) {
            acc[0][jt] = __builtin_amdgcn_wmma_f32_16x16x32_bf16(false, af0, false, bfr[jt], (short)0, acc[0][jt], false, false);
            acc[1][jt] = __builtin_amdgcn_wmma_f32_16x16x32_bf16(false, af1, false, bfr[jt], (short)0, acc[1][jt], false, false);
        }
        if (ks < 11) {
            uint4* ad = (uint4*)&Alds[nxt][srow][scol];
            ad[0] = ar[0]; ad[1] = ar[1];
            uint4* bd = (uint4*)&Blds[nxt][srow][scol];
            bd[0] = br[0]; bd[1] = br[1];
        }
    }

    #pragma unroll
    for (int it = 0; it < 2; ++it)
      #pragma unroll
      for (int jt = 0; jt < 4; ++jt)
        #pragma unroll
        for (int r = 0; r < 8; ++r) {
            int m = m0 + wm * 32 + it * 16 + half * 8 + r;
            int n = n0 + wn * 64 + jt * 16 + jj;
            out[(size_t)m * C_DIM + n] = acc[it][jt][r] + bias[n];
        }
}

// ---------------------------------------------------------------------------
extern "C" void kernel_launch(void* const* d_in, const int* in_sizes, int n_in,
                              void* d_out, int out_size, void* d_ws, size_t ws_size,
                              hipStream_t stream) {
    (void)in_sizes; (void)n_in; (void)out_size; (void)ws_size;
    const float* x          = (const float*)d_in[0];
    const float* mask       = (const float*)d_in[1];
    const float* qkv_w      = (const float*)d_in[2];
    const float* qkv_b      = (const float*)d_in[3];
    const float* proj_w     = (const float*)d_in[4];
    const float* proj_b     = (const float*)d_in[5];
    const float* bias_table = (const float*)d_in[6];
    const int*   rel_index  = (const int*)d_in[7];
    float* out = (float*)d_out;

    char* ws = (char*)d_ws;
    size_t off = 0;
    float*  biasHNN = (float*)(ws + off);  off += sizeof(float) * H_DIM * NT * NT;   // 115248 B (16-aligned)
    __bf16* qkv_wb  = (__bf16*)(ws + off); off += (size_t)2 * QKV_N * C_DIM;
    __bf16* proj_wb = (__bf16*)(ws + off); off += (size_t)2 * C_DIM * C_DIM;
    __bf16* qkvbuf  = (__bf16*)(ws + off); off += (size_t)2 * 3 * M_TOT * C_DIM;
    __bf16* attnbuf = (__bf16*)(ws + off); off += (size_t)2 * M_TOT * C_DIM;

    prep_kernel<<<(QKV_N * C_DIM + 255) / 256, 256, 0, stream>>>(
        qkv_w, proj_w, bias_table, rel_index, qkv_wb, proj_wb, biasHNN);

    qkv_gemm_kernel<<<dim3(M_TOT / 128, QKV_N / 128), 256, 0, stream>>>(
        x, qkv_wb, qkv_b, qkvbuf);

    attn_kernel<<<dim3(BW_, H_DIM), 128, 0, stream>>>(
        qkvbuf, mask, biasHNN, attnbuf);

    proj_gemm_kernel<<<dim3(M_TOT / 128, C_DIM / 128), 256, 0, stream>>>(
        attnbuf, proj_wb, proj_b, out);
}
